// EVRPTW_Model_69131793596747
// MI455X (gfx1250) — matrix operand order repaired and checked by reading, not compiled, not run on hardware
//
#include <hip/hip_runtime.h>
#include <hip/hip_bf16.h>
#include <math.h>

// ---------------------------------------------------------------------------
// EVRPTW attention decoder step for MI455X (gfx1250, wave32, WMMA).
// Dominant work: kvl = embedding @ W_kvl  (262144 x 256 x 768 GEMM, ~103 GFLOP)
// -> v_wmma_f32_16x16x32_bf16, fp32 accumulate, bf16 operands.
// embedding pre-converted to bf16 (fused with graph-mean pass); GEMM waves
// compute 2 M-tiles x 4 N-tiles to amortize fragment loads; kvl uses
// non-temporal stores/loads (384 MB stream, keeps 128 MB ebf L2-resident).
// ---------------------------------------------------------------------------

#define Bb   256
#define Ss   1024
#define Dd   256
#define Aa   256
#define Hh   8
#define DH   32
#define N3A  768   // 3*A

typedef __bf16 v16bf __attribute__((ext_vector_type(16)));
typedef float  v8f   __attribute__((ext_vector_type(8)));
typedef unsigned int uiv4 __attribute__((ext_vector_type(4)));  // native vec for NT ld

union Frag16 { uint4 q[2]; v16bf v; };   // 8 VGPRs = 16 bf16

// float -> bf16 (round-to-nearest-even), bf16 -> float via bit tricks
static __device__ __forceinline__ unsigned short f2bf(float f) {
  unsigned int u = __float_as_uint(f);
  u = (u + 0x7FFFu + ((u >> 16) & 1u)) >> 16;
  return (unsigned short)u;
}
static __device__ __forceinline__ float bf2f(unsigned short h) {
  return __uint_as_float(((unsigned int)h) << 16);
}
static __device__ __forceinline__ float bflo(unsigned int u) { return __uint_as_float(u << 16); }
static __device__ __forceinline__ float bfhi(unsigned int u) { return __uint_as_float(u & 0xFFFF0000u); }

// --------------------------------------------------------------------------
// K0: W_kvl [K=256, N=768] fp32  ->  Wt [N=768][K=256] bf16 (K-contiguous)
// --------------------------------------------------------------------------
__global__ void k_wt(const float* __restrict__ Wkvl, unsigned short* __restrict__ Wt) {
  int n = blockIdx.x;        // 0..767
  int k = threadIdx.x;       // 0..255
  Wt[(size_t)n * Dd + k] = f2bf(Wkvl[(size_t)k * N3A + n]);
}

// --------------------------------------------------------------------------
// K1: fused pass over embedding (one 256 MB read, NT):
//     ebf[b,s,d] = bf16(embedding[b,s,d]);  graph[b,d] = mean_s embedding
//     (ebf stored with default policy: it should stay L2-resident for K3)
// --------------------------------------------------------------------------
__global__ __launch_bounds__(256)
void k_ebf_mean(const float* __restrict__ emb, unsigned short* __restrict__ ebf,
                float* __restrict__ graph) {
  int b = blockIdx.x, d = threadIdx.x;
  const float* p = emb + (size_t)b * Ss * Dd + d;
  unsigned short* q = ebf + (size_t)b * Ss * Dd + d;
  float s = 0.0f;
  #pragma unroll 4
  for (int i = 0; i < Ss; ++i) {
    float v = __builtin_nontemporal_load(p + (size_t)i * Dd);
    s += v;
    q[(size_t)i * Dd] = f2bf(v);
  }
  graph[b * Dd + d] = s * (1.0f / (float)Ss);
}

// --------------------------------------------------------------------------
// K2: query[b,a] = graph[b,:]@W_ctx[:,a] + cur[b,:]@W_step[0:256,a]
//                 + (1 - used_cap[b]) * W_step[256,a]
// --------------------------------------------------------------------------
__global__ __launch_bounds__(256)
void k_query(const float* __restrict__ emb, const int* __restrict__ cn,
             const float* __restrict__ ucap, const float* __restrict__ graph,
             const float* __restrict__ Wctx, const float* __restrict__ Wstep,
             float* __restrict__ query) {
  __shared__ float g[Dd];
  __shared__ float c[Dd];
  int b = blockIdx.x, a = threadIdx.x;
  int node = cn[b];
  g[a] = graph[b * Dd + a];
  c[a] = emb[((size_t)b * Ss + node) * Dd + a];
  __syncthreads();
  float s = (1.0f - ucap[b]) * Wstep[Dd * Aa + a];
  #pragma unroll 4
  for (int d = 0; d < Dd; ++d) s = fmaf(g[d], Wctx[d * Aa + a], s);
  #pragma unroll 4
  for (int d = 0; d < Dd; ++d) s = fmaf(c[d], Wstep[d * Aa + a], s);
  query[b * Aa + a] = s;
}

// --------------------------------------------------------------------------
// K3: kvl[m, n] = ebf[m, :] @ Wt[n, :]  via v_wmma_f32_16x16x32_bf16
//     + time embedding fused into gk (n<256) and gv (256<=n<512), bf16 out.
// Grid: 8192 blocks (32-row M tiles), 12 waves/block.
// Each wave: 2 M-tiles x 4 N-tiles (64 cols) -> per K-step:
//   4 A b128-loads + 8 B b128-loads feed 8 WMMAs.
// A frag (16x32 bf16): lane m=lane&15; lanes 0-15 hold K 0..7 & 16..23,
//   lanes 16-31 hold K 8..15 & 24..31 -> two b128 loads from bf16 row-major.
// B frag (32x16 bf16): lane n=lane&15; same K split as A (0..7 & 16..23,
//   lanes 16-31: 8..15 & 24..31) -> two b128 loads from K-contiguous Wt.
// --------------------------------------------------------------------------
__global__ __launch_bounds__(384)
void k_kvl_gemm(const unsigned short* __restrict__ ebf,
                const unsigned short* __restrict__ Wt,
                const float* __restrict__ ctime, const float* __restrict__ Wtime,
                unsigned short* __restrict__ kvl) {
  const int lane = threadIdx.x & 31;
  const int wave = threadIdx.x >> 5;        // 0..11
  const int rowBase = blockIdx.x * 32;      // 32 consecutive rows, same batch b
  const int mrow = lane & 15;
  const int lhi  = lane >> 4;               // 0 or 1
  const int nbase = wave * 64;              // 64 cols per wave, 12*64 = 768

  v8f acc[8];                               // acc[mt*4 + nt]
  #pragma unroll
  for (int i = 0; i < 8; ++i) acc[i] = (v8f){0, 0, 0, 0, 0, 0, 0, 0};

  const unsigned short* arow0 = ebf + (size_t)(rowBase + mrow) * Dd + lhi * 8;
  const unsigned short* arow1 = arow0 + (size_t)16 * Dd;

  #pragma unroll
  for (int k0 = 0; k0 < Dd; k0 += 32) {
    // ---- A fragments for the two 16-row M tiles ----
    const uint4* ap0 = (const uint4*)(arow0 + k0);
    const uint4* ap1 = (const uint4*)(arow1 + k0);
    Frag16 a0, a1;
    a0.q[0] = ap0[0];  a0.q[1] = ap0[2];   // K ksel..+7, ksel+16..+23
    a1.q[0] = ap1[0];  a1.q[1] = ap1[2];

    #pragma unroll
    for (int nt = 0; nt < 4; ++nt) {
      int n = nbase + nt * 16 + mrow;
      Frag16 bf;
      const uint4* bp = (const uint4*)(Wt + (size_t)n * Dd + k0 + lhi * 8);
      bf.q[0] = bp[0];
      bf.q[1] = bp[2];
      acc[nt] = __builtin_amdgcn_wmma_f32_16x16x32_bf16(
          false, a0.v, false, bf.v, (short)0, acc[nt], false, false);
      acc[4 + nt] = __builtin_amdgcn_wmma_f32_16x16x32_bf16(
          false, a1.v, false, bf.v, (short)0, acc[4 + nt], false, false);
    }
  }

  // ---- epilogue: fused time-embedding add + NT bf16 store ----
  // C/D layout: VGPR r -> M = lhi*8 + r; N = lane&15 (+tile base).
  int b = rowBase >> 10;       // rowBase / 1024 (32 | 1024 so one batch/block)
  float ct = ctime[b];
  #pragma unroll
  for (int mt = 0; mt < 2; ++mt) {
    #pragma unroll
    for (int nt = 0; nt < 4; ++nt) {
      int n = nbase + nt * 16 + mrow;
      float te = (n < 512) ? ct * Wtime[n & 31] : 0.0f;
      #pragma unroll
      for (int r = 0; r < 8; ++r) {
        int row = rowBase + mt * 16 + lhi * 8 + r;
        __builtin_nontemporal_store(f2bf(acc[mt * 4 + nt][r] + te),
                                    kvl + (size_t)row * N3A + n);
      }
    }
  }
}

// --------------------------------------------------------------------------
// K4: single-query multi-head attention, one block per (b,h).
// kvl (gk/gv) is streamed once -> NT loads.
// --------------------------------------------------------------------------
__global__ __launch_bounds__(256)
void k_attn(const unsigned short* __restrict__ kvl, const float* __restrict__ query,
            const unsigned char* __restrict__ mask, float* __restrict__ heads) {
  __shared__ float qs[DH];
  __shared__ float sc[Ss];
  __shared__ float red[256];
  __shared__ float part[256];
  int b = blockIdx.x >> 3, h = blockIdx.x & 7;
  int t = threadIdx.x;
  if (t < DH) qs[t] = query[b * Aa + h * DH + t];
  __syncthreads();

  const float scale = 0.1767766952966369f; // 1/sqrt(32)
  float lmax = -INFINITY;
  for (int i = 0; i < 4; ++i) {
    int s = t + i * 256;
    const uiv4* kp = (const uiv4*)(kvl + ((size_t)(b * Ss + s)) * N3A + h * DH);
    float dot = 0.0f;
    #pragma unroll
    for (int c = 0; c < 4; ++c) {      // 4 * 8 bf16 = 32
      uiv4 u = __builtin_nontemporal_load(kp + c);
      dot += qs[c*8+0]*bflo(u.x) + qs[c*8+1]*bfhi(u.x)
           + qs[c*8+2]*bflo(u.y) + qs[c*8+3]*bfhi(u.y)
           + qs[c*8+4]*bflo(u.z) + qs[c*8+5]*bfhi(u.z)
           + qs[c*8+6]*bflo(u.w) + qs[c*8+7]*bfhi(u.w);
    }
    float cv = dot * scale;
    if (mask[b * Ss + s]) cv = -INFINITY;
    sc[s] = cv;
    lmax = fmaxf(lmax, cv);
  }
  red[t] = lmax; __syncthreads();
  for (int off = 128; off > 0; off >>= 1) {
    if (t < off) red[t] = fmaxf(red[t], red[t + off]);
    __syncthreads();
  }
  float m = red[0]; __syncthreads();

  float lsum = 0.0f;
  for (int i = 0; i < 4; ++i) {
    int s = t + i * 256;
    float e = __expf(sc[s] - m);
    sc[s] = e;
    lsum += e;
  }
  red[t] = lsum; __syncthreads();
  for (int off = 128; off > 0; off >>= 1) {
    if (t < off) red[t] += red[t + off];
    __syncthreads();
  }
  float inv = 1.0f / red[0];

  // weighted sum over V (gv region starts at column 256)
  int dh = t & 31, chunk = t >> 5;
  float p = 0.0f;
  for (int s = chunk * 128; s < chunk * 128 + 128; ++s) {
    unsigned short hv = __builtin_nontemporal_load(
        kvl + ((size_t)(b * Ss + s)) * N3A + Aa + h * DH + dh);
    p += sc[s] * bf2f(hv);
  }
  part[t] = p; __syncthreads();
  if (t < DH) {
    float sum = 0.0f;
    #pragma unroll
    for (int c = 0; c < 8; ++c) sum += part[c * 32 + t];
    heads[b * Aa + h * DH + t] = sum * inv;
  }
}

// --------------------------------------------------------------------------
// K5: glimpse[b,a] = heads[b,:] @ W_out[:,a]
// --------------------------------------------------------------------------
__global__ __launch_bounds__(256)
void k_glimpse(const float* __restrict__ heads, const float* __restrict__ Wout,
               float* __restrict__ glimpse) {
  __shared__ float hrow[Aa];
  int b = blockIdx.x, a = threadIdx.x;
  hrow[a] = heads[b * Aa + a];
  __syncthreads();
  float s = 0.0f;
  #pragma unroll 4
  for (int k = 0; k < Aa; ++k) s = fmaf(hrow[k], Wout[k * Aa + a], s);
  glimpse[b * Aa + a] = s;
}

// --------------------------------------------------------------------------
// K6: logits = tanh(glimpse . lk / 16) * 10, mask, log_softmax over S
// lk region of kvl is streamed once -> NT loads.
// --------------------------------------------------------------------------
__global__ __launch_bounds__(256)
void k_logits(const unsigned short* __restrict__ kvl, const float* __restrict__ glimpse,
              const unsigned char* __restrict__ mask, float* __restrict__ out) {
  __shared__ float g[Aa];
  __shared__ float lg[Ss];
  __shared__ float red[256];
  int b = blockIdx.x, t = threadIdx.x;
  g[t] = glimpse[b * Aa + t];
  __syncthreads();

  float lmax = -INFINITY;
  for (int i = 0; i < 4; ++i) {
    int s = t + i * 256;
    const uiv4* lp = (const uiv4*)(kvl + ((size_t)(b * Ss + s)) * N3A + 2 * Aa);
    float dot = 0.0f;
    #pragma unroll
    for (int c = 0; c < 32; ++c) {     // 32 * 8 bf16 = 256
      uiv4 u = __builtin_nontemporal_load(lp + c);
      dot += g[c*8+0]*bflo(u.x) + g[c*8+1]*bfhi(u.x)
           + g[c*8+2]*bflo(u.y) + g[c*8+3]*bfhi(u.y)
           + g[c*8+4]*bflo(u.z) + g[c*8+5]*bfhi(u.z)
           + g[c*8+6]*bflo(u.w) + g[c*8+7]*bfhi(u.w);
    }
    float l = tanhf(dot * (1.0f / 16.0f)) * 10.0f;   // TANH_CLIP = 10
    if (mask[b * Ss + s]) l = -INFINITY;
    lg[s] = l;
    lmax = fmaxf(lmax, l);
  }
  red[t] = lmax; __syncthreads();
  for (int off = 128; off > 0; off >>= 1) {
    if (t < off) red[t] = fmaxf(red[t], red[t + off]);
    __syncthreads();
  }
  float m = red[0]; __syncthreads();

  float lsum = 0.0f;
  for (int i = 0; i < 4; ++i) { int s = t + i * 256; lsum += __expf(lg[s] - m); }
  red[t] = lsum; __syncthreads();
  for (int off = 128; off > 0; off >>= 1) {
    if (t < off) red[t] += red[t + off];
    __syncthreads();
  }
  float lse = m + __logf(red[0]);
  for (int i = 0; i < 4; ++i) { int s = t + i * 256; out[b * Ss + s] = lg[s] - lse; }
}

// --------------------------------------------------------------------------
// Launch
// --------------------------------------------------------------------------
extern "C" void kernel_launch(void* const* d_in, const int* in_sizes, int n_in,
                              void* d_out, int out_size, void* d_ws, size_t ws_size,
                              hipStream_t stream) {
  (void)in_sizes; (void)n_in; (void)out_size; (void)ws_size;

  const float*         emb   = (const float*)d_in[0];          // [256,1024,256]
  const int*           cn    = (const int*)d_in[1];            // [256,1]
  const float*         ucap  = (const float*)d_in[2];          // [256,1]
  const float*         ctime = (const float*)d_in[3];          // [256,1,1]
  const unsigned char* mask  = (const unsigned char*)d_in[4];  // [256,1,1024] bool (1B)
  const float*         Wctx  = (const float*)d_in[5];          // [256,256]
  const float*         Wkvl  = (const float*)d_in[6];          // [256,768]
  const float*         Wstep = (const float*)d_in[7];          // [257,256]
  const float*         Wout  = (const float*)d_in[8];          // [256,256]
  const float*         Wtime = (const float*)d_in[9];          // [1,32]
  float*               out   = (float*)d_out;                  // [256,1,1024]

  // workspace layout
  char* ws = (char*)d_ws;
  unsigned short* Wt      = (unsigned short*)ws; ws += (size_t)N3A * Dd * 2;   // 384 KB
  float*          graph   = (float*)ws;          ws += (size_t)Bb * Dd * 4;
  float*          query   = (float*)ws;          ws += (size_t)Bb * Aa * 4;
  float*          heads   = (float*)ws;          ws += (size_t)Bb * Aa * 4;
  float*          glimpse = (float*)ws;          ws += (size_t)Bb * Aa * 4;
  unsigned short* ebf     = (unsigned short*)ws; ws += (size_t)Bb * Ss * Dd * 2; // 128 MB
  unsigned short* kvl     = (unsigned short*)ws; // [B*S, 768] bf16 = 384 MB

  k_wt        <<<N3A,  256, 0, stream>>>(Wkvl, Wt);
  k_ebf_mean  <<<Bb,   256, 0, stream>>>(emb, ebf, graph);
  k_query     <<<Bb,   256, 0, stream>>>(emb, cn, ucap, graph, Wctx, Wstep, query);
  k_kvl_gemm  <<<(Bb * Ss) / 32, 384, 0, stream>>>(ebf, Wt, ctime, Wtime, kvl);
  k_attn      <<<Bb * Hh, 256, 0, stream>>>(kvl, query, mask, heads);
  k_glimpse   <<<Bb,   256, 0, stream>>>(heads, Wout, glimpse);
  k_logits    <<<Bb,   256, 0, stream>>>(kvl, glimpse, mask, out);
}
